// HalfKPNetwork_39840116637857
// MI455X (gfx1250) — matrix operand order repaired and checked by reading, not compiled, not run on hardware
//
#include <hip/hip_runtime.h>
#include <hip/hip_bf16.h>
#include <math.h>

typedef __attribute__((ext_vector_type(16))) _Float16    v16h;
typedef __attribute__((ext_vector_type(8)))  float       v8f;
typedef __attribute__((ext_vector_type(8)))  unsigned int v8u;

#define FDIM  40960
#define BATCH 2048
#define H1    256
#define KC    256
#define NK32  (FDIM / 32)   /* 1280 k-steps of 32 */

// pack two f32 into one dword of two f16 (v_cvt_pk_rtz_f16_f32)
static __device__ __forceinline__ unsigned pk(float a, float b) {
  auto t = __builtin_amdgcn_cvt_pkrtz(a, b);   // __fp16 ext_vector(2)
  return __builtin_bit_cast(unsigned, t);
}

// assemble a 16-element f16 fragment (8 VGPRs) from two 128-bit loads
static __device__ __forceinline__ v16h frag(uint4 lo, uint4 hi) {
  v8u r = {lo.x, lo.y, lo.z, lo.w, hi.x, hi.y, hi.z, hi.w};
  return __builtin_bit_cast(v16h, r);
}

static __device__ __forceinline__ v8f wmma_f16(v16h a, v16h b, v8f c) {
  // (neg_a, A, neg_b, B, c_mod, C, reuse_a, reuse_b)
  return __builtin_amdgcn_wmma_f32_16x16x32_f16(false, a, false, b, (short)0, c, false, false);
}

// ---------------------------------------------------------------------------
// Kernel 1: prepack W_fw / W_fb (f32, [256][40960] row-major) into f16
// B-fragment layout: per (side,ntile,kb32) tile: 32 lanes x 8 dwords.
//   lane -> N = lane%16 ; dword v -> K pair kb32*32 + (lane/16)*8 + 2v + (v/4)*8 (+0,+1), even K in [15:0]
// One wave per tile; 2*16*1280 = 40960 tiles.
// ---------------------------------------------------------------------------
__global__ __launch_bounds__(256) void prepack_w(
    const float* __restrict__ Wfw, const float* __restrict__ Wfb,
    unsigned* __restrict__ Wpk)
{
  const int gw    = blockIdx.x * 8 + (threadIdx.x >> 5);
  const int lane  = threadIdx.x & 31;
  const int kb32  = gw % NK32;
  const int t     = gw / NK32;        // side*16 + ntile
  const int ntile = t & 15;
  const float* W  = (t >> 4) ? Wfb : Wfw;
  const float* wp = W + (size_t)(ntile * 16 + (lane & 15)) * FDIM + kb32 * 32 + (lane >> 4) * 8;
  unsigned* dst   = Wpk + ((size_t)t * NK32 + kb32) * 256 + lane * 8;
#pragma unroll
  for (int v = 0; v < 8; ++v)
    dst[v] = pk(wp[2 * v + (v >> 2) * 8], wp[2 * v + 1 + (v >> 2) * 8]);
}

// ---------------------------------------------------------------------------
// Kernel 2: feature transformer GEMM + bias + ReLU
//   xbuf[b][0:256]   = relu(white @ W_fw^T + b_fw)
//   xbuf[b][256:512] = relu(black @ W_fb^T + b_fb)
// Block: 512 threads = 16 waves; M-tile of 16 rows, each wave owns one 16-col
// N-tile. K streamed in KC=256 chunks through double-buffered LDS (f16 packed)
// with register pipelining: load chunk i+1 while computing chunk i; a single
// workgroup barrier per chunk.
// ---------------------------------------------------------------------------
#define LDSB (16 * (KC / 2))   /* dwords per LDS buffer: 2048 */

__global__ __launch_bounds__(512) void ft_gemm(
    const float* __restrict__ Xw, const float* __restrict__ Xb,
    const float* __restrict__ bw, const float* __restrict__ bb,
    const unsigned* __restrict__ Wpk, float* __restrict__ xbuf)
{
  __shared__ unsigned lds[2 * LDSB];           // 2 x (16 rows x 256 f16) = 16 KB

  const int m0   = blockIdx.x * 16;
  const int side = blockIdx.y;                 // 0 = white, 1 = black
  const int tid  = threadIdx.x;
  const int lane = tid & 31;
  const int wave = tid >> 5;                   // 0..15 : ntile and staging row
  const int half = lane >> 4;
  const int mrow = lane & 15;

  const float* X    = side ? Xb : Xw;
  const float* bias = side ? bb : bw;
  const unsigned* wb =
      Wpk + ((size_t)(side * 16 + wave) * NK32) * 256 + lane * 8;

  // staging: each wave stages its own row; lane covers 8 floats (2 x float4)
  const float* xrow = X + (size_t)(m0 + wave) * FDIM + lane * 4;
  const int stoff = wave * (KC / 2) + lane * 2;   // dword offset within a buffer

  const int NCHUNK = FDIM / KC;                // 160

  // ---- preload + stage chunk 0 into buffer 0 ----
  float4 xa = *(const float4*)(xrow);
  float4 xc = *(const float4*)(xrow + 128);
  {
    uint2 p0, p1;
    p0.x = pk(xa.x, xa.y); p0.y = pk(xa.z, xa.w);
    p1.x = pk(xc.x, xc.y); p1.y = pk(xc.z, xc.w);
    *(uint2*)&lds[stoff]      = p0;
    *(uint2*)&lds[stoff + 64] = p1;
  }
  __syncthreads();

  v8f acc = {};
  for (int i = 0; i < NCHUNK; ++i) {
    const int kb = i * KC;

    // ---- issue loads for chunk i+1 (latency hidden under compute) ----
    float4 na, nc;
    const bool more = (i + 1 < NCHUNK);
    if (more) {
      na = *(const float4*)(xrow + kb + KC);
      nc = *(const float4*)(xrow + kb + KC + 128);
      if (kb + 2 * KC < FDIM) __builtin_prefetch(xrow + kb + 2 * KC, 0, 1);
    }

    // ---- compute chunk i from lds buffer (i&1) ----
    const unsigned* buf = &lds[(i & 1) * LDSB];
    const int kb32 = kb >> 5;
#pragma unroll
    for (int s = 0; s < KC / 32; ++s) {
      // A fragment: lane (M=mrow) reads K runs {kk+half*8..+7, kk+16+half*8..+7}
      const unsigned* ab = &buf[mrow * (KC / 2) + s * 16 + half * 4];
      uint4 alo = *(const uint4*)(ab);
      uint4 ahi = *(const uint4*)(ab + 8);
      // B fragment: prepacked f16, 2 coalesced b128 per lane
      const uint4* bp = (const uint4*)(wb + (size_t)(kb32 + s) * 256);
      uint4 blo = bp[0];
      uint4 bhi = bp[1];
      acc = wmma_f16(frag(alo, ahi), frag(blo, bhi), acc);
    }

    // ---- stage chunk i+1 into the other buffer ----
    if (more) {
      unsigned* dbuf = &lds[((i + 1) & 1) * LDSB];
      uint2 p0, p1;
      p0.x = pk(na.x, na.y); p0.y = pk(na.z, na.w);
      p1.x = pk(nc.x, nc.y); p1.y = pk(nc.z, nc.w);
      *(uint2*)&dbuf[stoff]      = p0;
      *(uint2*)&dbuf[stoff + 64] = p1;
    }
    __syncthreads();
  }

  // epilogue: C layout -> VGPR i holds (M = i + 8*half, N = mrow)
  const int ncol = wave * 16 + mrow;           // 0..255 within this side
  const float bv = bias[ncol];
  float* op = xbuf + (size_t)(m0 + half * 8) * (2 * H1) + side * H1 + ncol;
#pragma unroll
  for (int i = 0; i < 8; ++i) {
    float v = acc[i] + bv;
    op[(size_t)i * (2 * H1)] = v > 0.f ? v : 0.f;
  }
}

// ---------------------------------------------------------------------------
// Kernel 3: tail  x[16][512] -> relu(W1) -> relu(W2) -> tanh(W3)
// One block (2 waves) per 16 batch rows; WMMA for layers 1 and 2.
// ---------------------------------------------------------------------------
__global__ __launch_bounds__(64) void tail_net(
    const float* __restrict__ xbuf,
    const float* __restrict__ W1, const float* __restrict__ b1,
    const float* __restrict__ W2, const float* __restrict__ b2,
    const float* __restrict__ W3, const float* __restrict__ b3,
    float* __restrict__ out)
{
  __shared__ float h1s[16 * 32];
  __shared__ float h2s[16 * 32];

  const int m0   = blockIdx.x * 16;
  const int tid  = threadIdx.x;
  const int lane = tid & 31;
  const int wave = tid >> 5;                   // 0..1 : N-tile of layer outputs
  const int half = lane >> 4;
  const int mrow = lane & 15;
  const int ncol = wave * 16 + mrow;

  // ---- layer 1: [16x512] @ [512x32] ----
  v8f acc = {};
  for (int kb = 0; kb < 2 * H1; kb += 32) {
    const float* ap = xbuf + (size_t)(m0 + mrow) * (2 * H1) + kb + half * 8;
    float4 a0 = *(const float4*)(ap);
    float4 a1 = *(const float4*)(ap + 4);
    float4 a2 = *(const float4*)(ap + 16);
    float4 a3 = *(const float4*)(ap + 20);
    uint4 alo = {pk(a0.x, a0.y), pk(a0.z, a0.w), pk(a1.x, a1.y), pk(a1.z, a1.w)};
    uint4 ahi = {pk(a2.x, a2.y), pk(a2.z, a2.w), pk(a3.x, a3.y), pk(a3.z, a3.w)};

    const float* wp = W1 + (size_t)(wave * 16 + mrow) * (2 * H1) + kb + half * 8; // N = ncol, K table order
    unsigned bwd[8];
#pragma unroll
    for (int v = 0; v < 8; ++v)
      bwd[v] = pk(wp[2 * v + (v >> 2) * 8],
                  wp[2 * v + 1 + (v >> 2) * 8]);
    uint4 blo = {bwd[0], bwd[1], bwd[2], bwd[3]};
    uint4 bhi = {bwd[4], bwd[5], bwd[6], bwd[7]};
    acc = wmma_f16(frag(alo, ahi), frag(blo, bhi), acc);
  }
  {
    const float bv = b1[ncol];
#pragma unroll
    for (int i = 0; i < 8; ++i) {
      float v = acc[i] + bv;
      h1s[(half * 8 + i) * 32 + ncol] = v > 0.f ? v : 0.f;
    }
  }
  __syncthreads();

  // ---- layer 2: [16x32] @ [32x32], single WMMA per wave ----
  v8f acc2 = {};
  {
    const float* ap = &h1s[mrow * 32 + half * 8];
    float4 a0 = *(const float4*)(ap);
    float4 a1 = *(const float4*)(ap + 4);
    float4 a2 = *(const float4*)(ap + 16);
    float4 a3 = *(const float4*)(ap + 20);
    uint4 alo = {pk(a0.x, a0.y), pk(a0.z, a0.w), pk(a1.x, a1.y), pk(a1.z, a1.w)};
    uint4 ahi = {pk(a2.x, a2.y), pk(a2.z, a2.w), pk(a3.x, a3.y), pk(a3.z, a3.w)};

    const float* wp = W2 + (size_t)(wave * 16 + mrow) * 32 + half * 8; // N = ncol, K table order
    unsigned bwd[8];
#pragma unroll
    for (int v = 0; v < 8; ++v)
      bwd[v] = pk(wp[2 * v + (v >> 2) * 8], wp[2 * v + 1 + (v >> 2) * 8]);
    uint4 blo = {bwd[0], bwd[1], bwd[2], bwd[3]};
    uint4 bhi = {bwd[4], bwd[5], bwd[6], bwd[7]};
    acc2 = wmma_f16(frag(alo, ahi), frag(blo, bhi), acc2);
  }
  {
    const float bv = b2[ncol];
#pragma unroll
    for (int i = 0; i < 8; ++i) {
      float v = acc2[i] + bv;
      h2s[(half * 8 + i) * 32 + ncol] = v > 0.f ? v : 0.f;
    }
  }
  __syncthreads();

  // ---- layer 3 + tanh: one row per lane ----
  if (tid < 16) {
    float s = b3[0];
#pragma unroll
    for (int k = 0; k < 32; ++k) s += h2s[tid * 32 + k] * W3[k];
    out[m0 + tid] = tanhf(s);
  }
}

// ---------------------------------------------------------------------------
// launch
// d_in order: white, black, W_fw, b_fw, W_fb, b_fb, W1, b1, W2, b2, W3, b3
// d_ws: [0, 41.9MB) packed f16 weights ; [41.9MB, +4MB) xbuf [2048][512] f32
// ---------------------------------------------------------------------------
extern "C" void kernel_launch(void* const* d_in, const int* in_sizes, int n_in,
                              void* d_out, int out_size, void* d_ws, size_t ws_size,
                              hipStream_t stream) {
  const float* Xw  = (const float*)d_in[0];
  const float* Xb  = (const float*)d_in[1];
  const float* Wfw = (const float*)d_in[2];
  const float* bfw = (const float*)d_in[3];
  const float* Wfb = (const float*)d_in[4];
  const float* bfb = (const float*)d_in[5];
  const float* W1  = (const float*)d_in[6];
  const float* b1  = (const float*)d_in[7];
  const float* W2  = (const float*)d_in[8];
  const float* b2  = (const float*)d_in[9];
  const float* W3  = (const float*)d_in[10];
  const float* b3  = (const float*)d_in[11];
  float* out = (float*)d_out;

  unsigned* Wpk = (unsigned*)d_ws;
  const size_t wpk_bytes = (size_t)2 * 16 * NK32 * 256 * sizeof(unsigned); // 41.9 MB
  float* xbuf = (float*)((char*)d_ws + wpk_bytes);                         // 4 MB

  prepack_w<<<dim3((2 * 16 * NK32) / 8), 256, 0, stream>>>(Wfw, Wfb, Wpk);
  ft_gemm<<<dim3(BATCH / 16, 2), 512, 0, stream>>>(Xw, Xb, bfw, bfb, Wpk, xbuf);
  tail_net<<<dim3(BATCH / 16), 64, 0, stream>>>(xbuf, W1, b1, W2, b2, W3, b3, out);
}
